// RCNModel_41781441855935
// MI455X (gfx1250) — matrix operand order repaired and checked by reading, not compiled, not run on hardware
//
#include <hip/hip_runtime.h>
#include <math.h>
#include <stdint.h>

// ---------------------------------------------------------------------------
// Problem constants (from the reference)
// ---------------------------------------------------------------------------
#define NN      65536      // nodes
#define EE      262144     // edges
#define NG      1024       // graphs
#define NPG     64         // nodes per graph
#define EPG     256        // edges per graph
#define HH      4          // heads
#define CC      128        // channels per head
#define HC      512        // H*C
#define GAT_IN  67
#define GAT_INP 96         // padded K for GAT1
#define GDIM    72
#define GEMB    32
#define MLP_IN  160
#define POLICY  4672

// Input tensor indices (setup_inputs dict order, params depth-first insertion order)
enum {
  IN_NODE_TYPES = 0, IN_X_REST, IN_EDGE_INDEX, IN_EDGE_ATTR, IN_BATCH, IN_GLOBAL_FEATS,
  IN_EMB_NODE, IN_EMB_EDGE, IN_WG, IN_BG,
  IN_G1WL, IN_G1BL, IN_G1WR, IN_G1BR, IN_G1WE, IN_G1ATT, IN_G1BIAS,
  IN_GAMMA1, IN_BETA1,
  IN_G2WL, IN_G2BL, IN_G2WR, IN_G2BR, IN_G2WE, IN_G2ATT, IN_G2BIAS,
  IN_GAMMA2, IN_BETA2,
  IN_WV1, IN_BV1, IN_WV2, IN_BV2,
  IN_WP1, IN_BP1, IN_WP2, IN_BP2,
  IN_WT1, IN_BT1, IN_WT2, IN_BT2,
  IN_WS1, IN_BS1, IN_WS2, IN_BS2
};

typedef __attribute__((ext_vector_type(16))) _Float16 v16h;
typedef __attribute__((ext_vector_type(8)))  _Float16 v8h;
typedef __attribute__((ext_vector_type(8)))  float    v8f;

// ---------------------------------------------------------------------------
// WMMA GEMM: Out[M,N] = act(A[M,K](f16) @ W[K,N] + bias[N])
//   - A is f16 row-major [M,K]; weights pre-transposed+converted: WT[N,K] f16.
//   - Block tile 128x64, 256 threads = 8 waves (4 m-strips x 2 n-strips);
//     each wave owns a 32x32 tile = four v_wmma_f32_16x16x32_f16, reusing
//     each A/B fragment twice.
//   - Tiles staged global->LDS with global_load_async_to_lds_b128 (ASYNCcnt),
//     one 16B chunk per issue, then s_wait_asynccnt + workgroup barrier.
//   - LDS rows use a 40-half stride: the 16B fragment reads (ds_load_b128)
//     are bank-conflict-free.
// Requires M%128==0, N%64==0, K%32==0 (all call sites satisfy this).
// ---------------------------------------------------------------------------
#define LDSTR 40

__device__ __forceinline__ void async_cp16(unsigned lds_addr, const _Float16* g) {
  uint64_t ga = (uint64_t)(uintptr_t)g;
  asm volatile("global_load_async_to_lds_b128 %0, %1, off"
               :: "v"(lds_addr), "v"(ga) : "memory");
}

__device__ __forceinline__ v16h frag_a(const _Float16* row, int lhi) {
  // a[i] = row[8*lhi + (i<8 ? i : i+8)]   (ISA 16-bit A 16x32 layout)
  v8h lo = *(const v8h*)(row + lhi * 8);
  v8h hi = *(const v8h*)(row + lhi * 8 + 16);
  return __builtin_shufflevector(lo, hi, 0,1,2,3,4,5,6,7,8,9,10,11,12,13,14,15);
}
__device__ __forceinline__ v16h frag_b(const _Float16* row, int lhi) {
  // b[i] = row[8*lhi + (i<8 ? i : i+8)]   (same k order as A)
  v8h lo = *(const v8h*)(row + lhi * 8);
  v8h hi = *(const v8h*)(row + lhi * 8 + 16);
  return __builtin_shufflevector(lo, hi, 0,1,2,3,4,5,6,7,8,9,10,11,12,13,14,15);
}

__global__ __launch_bounds__(256)
void k_wmma_gemm(const _Float16* __restrict__ A, int lda,
                 const _Float16* __restrict__ WT, int ldwt,
                 const float* __restrict__ bias,
                 void* __restrict__ OutP, int ldo,
                 int M, int N, int K, int act, int outHalf) {
  __shared__ __align__(16) _Float16 Asub[128 * LDSTR];   // 10 KB
  __shared__ __align__(16) _Float16 Bsub[64 * LDSTR];    // 5 KB

  const int tid  = threadIdx.x;
  const int lane = tid & 31;
  const int wv   = tid >> 5;
  const int wm   = wv & 3;      // 0..3 : 32-row strip
  const int wn   = wv >> 2;     // 0..1 : 32-col strip
  const int m0   = blockIdx.y * 128;
  const int n0   = blockIdx.x * 64;
  const int lh   = lane & 15;
  const int lhi  = lane >> 4;

  // Per-thread staging chunks (16B each): A = 512 chunks, B = 256 chunks
  const int ar0 = tid >> 2,           ac0 = (tid & 3) << 3;          // A chunk 0
  const int ar1 = (tid + 256) >> 2,   ac1 = ac0;                     // A chunk 1
  const int br  = tid >> 2,           bc  = (tid & 3) << 3;          // B chunk
  const unsigned lA0 = (unsigned)(uintptr_t)(Asub + ar0 * LDSTR + ac0);
  const unsigned lA1 = (unsigned)(uintptr_t)(Asub + ar1 * LDSTR + ac1);
  const unsigned lB  = (unsigned)(uintptr_t)(Bsub + br  * LDSTR + bc);
  const _Float16* gA0 = A  + (size_t)(m0 + ar0) * lda  + ac0;
  const _Float16* gA1 = A  + (size_t)(m0 + ar1) * lda  + ac1;
  const _Float16* gB  = WT + (size_t)(n0 + br)  * ldwt + bc;

  v8f c00 = {}, c01 = {}, c10 = {}, c11 = {};

  const _Float16* arow0 = Asub + (wm * 32 + lh) * LDSTR;
  const _Float16* arow1 = arow0 + 16 * LDSTR;
  const _Float16* brow0 = Bsub + (wn * 32 + lh) * LDSTR;
  const _Float16* brow1 = brow0 + 16 * LDSTR;

  for (int k0 = 0; k0 < K; k0 += 32) {
    async_cp16(lA0, gA0 + k0);
    async_cp16(lA1, gA1 + k0);
    async_cp16(lB,  gB  + k0);
    asm volatile("s_wait_asynccnt 0x0" ::: "memory");
    __syncthreads();

    v16h a0 = frag_a(arow0, lhi);
    v16h a1 = frag_a(arow1, lhi);
    v16h b0 = frag_b(brow0, lhi);
    v16h b1 = frag_b(brow1, lhi);

    c00 = __builtin_amdgcn_wmma_f32_16x16x32_f16(false, a0, false, b0, (short)0, c00, false, false);
    c01 = __builtin_amdgcn_wmma_f32_16x16x32_f16(false, a0, false, b1, (short)0, c01, false, false);
    c10 = __builtin_amdgcn_wmma_f32_16x16x32_f16(false, a1, false, b0, (short)0, c10, false, false);
    c11 = __builtin_amdgcn_wmma_f32_16x16x32_f16(false, a1, false, b1, (short)0, c11, false, false);
    __syncthreads();
  }

  // C/D layout: d[r] -> (M = r + 8*lhi, N = lane%16)
  const int outn = n0 + wn * 32 + lh;
  const float bb0 = bias[outn];
  const float bb1 = bias[outn + 16];
  #pragma unroll
  for (int mt = 0; mt < 2; ++mt) {
    const v8f& ca = mt ? c10 : c00;
    const v8f& cb = mt ? c11 : c01;
    #pragma unroll
    for (int r = 0; r < 8; ++r) {
      int row = m0 + wm * 32 + mt * 16 + r + lhi * 8;
      float v0 = ca[r] + bb0;
      float v1 = cb[r] + bb1;
      if (act) { v0 = fmaxf(v0, 0.f); v1 = fmaxf(v1, 0.f); }
      if (outHalf) {
        _Float16* Out = (_Float16*)OutP;
        Out[(size_t)row * ldo + outn]      = (_Float16)v0;
        Out[(size_t)row * ldo + outn + 16] = (_Float16)v1;
      } else {
        float* Out = (float*)OutP;
        Out[(size_t)row * ldo + outn]      = v0;
        Out[(size_t)row * ldo + outn + 16] = v1;
      }
    }
  }
}

// ---------------------------------------------------------------------------
// Helpers / small kernels
// ---------------------------------------------------------------------------
__device__ __forceinline__ unsigned fflip(float f) {
  unsigned u = __float_as_uint(f);
  return u ^ ((u >> 31) ? 0xFFFFFFFFu : 0x80000000u);
}
__device__ __forceinline__ float funflip(unsigned u) {
  u ^= ((u >> 31) ? 0x80000000u : 0xFFFFFFFFu);
  return __uint_as_float(u);
}

__device__ __forceinline__ float block_sum(float v, float* sh, int tid) {
  #pragma unroll
  for (int o = 16; o > 0; o >>= 1) v += __shfl_down(v, o, 32);
  if ((tid & 31) == 0) sh[tid >> 5] = v;
  __syncthreads();
  if (tid == 0) {
    float r = 0.f;
    int nw = (int)(blockDim.x >> 5);
    for (int i = 0; i < nw; ++i) r += sh[i];
    sh[0] = r;
  }
  __syncthreads();
  float r = sh[0];
  __syncthreads();
  return r;
}

// x_pad[N,96] (f16) = concat(emb_node[nt], x_rest, zeros)
__global__ void k_build_xh(const int* __restrict__ nt, const float* __restrict__ xr,
                           const float* __restrict__ emb, _Float16* __restrict__ x) {
  int idx = blockIdx.x * 256 + threadIdx.x;
  if (idx >= NN * GAT_INP) return;
  int n = idx / GAT_INP, j = idx - n * GAT_INP;
  float v;
  if (j < 64)      v = emb[nt[n] * 64 + j];
  else if (j < 67) v = xr[n * 3 + (j - 64)];
  else             v = 0.f;
  x[idx] = (_Float16)v;
}

// WT[N,Kpad] (f16) = transpose(W[K,N]) with zero K-padding
__global__ void k_prep_wt(const float* __restrict__ W, int K, int Kpad, int N,
                          _Float16* __restrict__ WT) {
  int idx = blockIdx.x * 256 + threadIdx.x;
  if (idx >= N * Kpad) return;
  int n = idx / Kpad, k = idx - n * Kpad;
  WT[idx] = (k < K) ? (_Float16)W[(size_t)k * N + n] : (_Float16)0.f;
}

// eet[3,512] = emb_edge[3,16] @ We[16,512]  (only 3 distinct edge types)
__global__ void k_edge_tab(const float* __restrict__ emb_edge,
                           const float* __restrict__ We, float* __restrict__ eet) {
  int a = blockIdx.x, j = threadIdx.x;   // 3 blocks x 512 threads
  float s = 0.f;
  #pragma unroll
  for (int k = 0; k < 16; ++k) s += emb_edge[a * 16 + k] * We[k * HC + j];
  eet[a * HC + j] = s;
}

// gv[B,32] = relu(global_feats[B,72] @ Wg + bg)
__global__ void k_global_mlp(const float* __restrict__ gf, const float* __restrict__ Wg,
                             const float* __restrict__ bg, float* __restrict__ gv) {
  int g = blockIdx.x, j = threadIdx.x;   // 1024 blocks x 32 threads
  float s = bg[j];
  for (int k = 0; k < GDIM; ++k) s += gf[g * GDIM + k] * Wg[k * GEMB + j];
  gv[g * GEMB + j] = fmaxf(s, 0.f);
}

// s[e,h] = sum_c lrelu(xl[src]+xr[dst]+eet[attr], 0.2) * att[h,c]; one wave/edge
__global__ __launch_bounds__(256)
void k_edge_scores(const float* __restrict__ xl, const float* __restrict__ xr,
                   const float* __restrict__ eet, const int* __restrict__ src,
                   const int* __restrict__ dst, const int* __restrict__ eattr,
                   const float* __restrict__ att, float* __restrict__ s) {
  int lane = threadIdx.x & 31, wv = threadIdx.x >> 5;
  int e = blockIdx.x * 8 + wv;
  int si = src[e], di = dst[e], a = eattr[e];
  const float* pl = xl + (size_t)si * HC;
  const float* pr = xr + (size_t)di * HC;
  const float* pe = eet + (size_t)a * HC;
  #pragma unroll
  for (int h = 0; h < HH; ++h) {
    float acc = 0.f;
    #pragma unroll
    for (int q = 0; q < 4; ++q) {
      int c = h * CC + q * 32 + lane;
      float m = pl[c] + pr[c] + pe[c];
      float lr = (m > 0.f) ? m : 0.2f * m;
      acc += lr * att[h * CC + q * 32 + lane];
    }
    #pragma unroll
    for (int o = 16; o > 0; o >>= 1) acc += __shfl_down(acc, o, 32);
    if (lane == 0) s[(size_t)e * HH + h] = acc;
  }
}

// Per-graph softmax over incoming edges (segment max + sum entirely in LDS).
// s[e,h] replaced in-place by alpha[e,h].
__global__ __launch_bounds__(256)
void k_softmax_graph(float* __restrict__ s, const int* __restrict__ dst) {
  __shared__ unsigned smax[NPG * HH];
  __shared__ float    den[NPG * HH];
  int g = blockIdx.x, tid = threadIdx.x;   // 256 threads == EPG
  smax[tid] = 0u;                           // < fflip(any real): acts as -inf
  den[tid] = 0.f;
  __syncthreads();
  int e = g * EPG + tid;
  int ld = dst[e] - g * NPG;               // 0..63 by construction
  float sv[HH];
  #pragma unroll
  for (int h = 0; h < HH; ++h) {
    sv[h] = s[(size_t)e * HH + h];
    atomicMax(&smax[ld * HH + h], fflip(sv[h]));
  }
  __syncthreads();
  float ex[HH];
  #pragma unroll
  for (int h = 0; h < HH; ++h) {
    ex[h] = expf(sv[h] - funflip(smax[ld * HH + h]));
    atomicAdd(&den[ld * HH + h], ex[h]);
  }
  __syncthreads();
  #pragma unroll
  for (int h = 0; h < HH; ++h)
    s[(size_t)e * HH + h] = ex[h] / (den[ld * HH + h] + 1e-16f);
}

// out[dst, h*128+c] += xl[src, h*128+c] * alpha[e,h]; LDS ds_add_f32 per
// (graph, head) block — no global atomics.
__global__ __launch_bounds__(256)
void k_aggregate(const float* __restrict__ xl, const float* __restrict__ alpha,
                 const int* __restrict__ src, const int* __restrict__ dst,
                 float* __restrict__ out) {
  __shared__ float acc[NPG * CC];          // 32 KB
  int g = blockIdx.x, h = blockIdx.y;
  int tid = threadIdx.x, lane = tid & 31, wv = tid >> 5;
  for (int i = tid; i < NPG * CC; i += 256) acc[i] = 0.f;
  __syncthreads();
  int ebase = g * EPG + wv * 32;
  for (int it = 0; it < 32; ++it) {
    int e = ebase + it;
    float a = alpha[(size_t)e * HH + h];
    int ld = dst[e] - g * NPG;
    const float* p = xl + (size_t)src[e] * HC + h * CC;
    #pragma unroll
    for (int q = 0; q < 4; ++q) {
      int c = q * 32 + lane;
      atomicAdd(&acc[ld * CC + c], p[c] * a);
    }
  }
  __syncthreads();
  for (int i = tid; i < NPG * CC; i += 256) {
    int ln = i >> 7, c = i & 127;
    out[(size_t)(g * NPG + ln) * HC + h * CC + c] = acc[i];
  }
}

// x1(f16) = relu(layernorm(in + bias, g, b)), D = 512, one block per node
__global__ __launch_bounds__(256)
void k_ln_relu_512(const float* __restrict__ in, const float* __restrict__ bias,
                   const float* __restrict__ g, const float* __restrict__ b,
                   _Float16* __restrict__ out) {
  __shared__ float sh[8];
  int n = blockIdx.x, tid = threadIdx.x;
  const float* row = in + (size_t)n * HC;
  float v0 = row[tid] + bias[tid];
  float v1 = row[tid + 256] + bias[tid + 256];
  float mu = block_sum(v0 + v1, sh, tid) * (1.f / 512.f);
  float d0 = v0 - mu, d1 = v1 - mu;
  float var = block_sum(d0 * d0 + d1 * d1, sh, tid) * (1.f / 512.f);
  float inv = rsqrtf(var + 1e-5f);
  _Float16* orow = out + (size_t)n * HC;
  orow[tid]       = (_Float16)fmaxf(d0 * inv * g[tid] + b[tid], 0.f);
  orow[tid + 256] = (_Float16)fmaxf(d1 * inv * g[tid + 256] + b[tid + 256], 0.f);
}

// x2 = relu(layernorm(mean_heads(in) + bias, g, b)), D = 128
__global__ __launch_bounds__(128)
void k_ln2_relu(const float* __restrict__ in, const float* __restrict__ bias,
                const float* __restrict__ g, const float* __restrict__ b,
                float* __restrict__ out) {
  __shared__ float sh[4];
  int n = blockIdx.x, c = threadIdx.x;
  const float* row = in + (size_t)n * HC;
  float y = 0.25f * (row[c] + row[c + 128] + row[c + 256] + row[c + 384]) + bias[c];
  float mu = block_sum(y, sh, c) * (1.f / 128.f);
  float d = y - mu;
  float var = block_sum(d * d, sh, c) * (1.f / 128.f);
  out[(size_t)n * CC + c] = fmaxf(d * rsqrtf(var + 1e-5f) * g[c] + b[c], 0.f);
}

// comb[g] (f16) = concat(mean over 64 nodes of x2[g], gv[g])
__global__ __launch_bounds__(128)
void k_pool_concat(const float* __restrict__ x2, const float* __restrict__ gv,
                   _Float16* __restrict__ comb) {
  int g = blockIdx.x, c = threadIdx.x;
  float s = 0.f;
  for (int i = 0; i < NPG; ++i) s += x2[(size_t)(g * NPG + i) * CC + c];
  comb[(size_t)g * MLP_IN + c] = (_Float16)(s * (1.f / 64.f));
  if (c < GEMB) comb[(size_t)g * MLP_IN + CC + c] = (_Float16)gv[g * GEMB + c];
}

// out[row] = act(hin[row,:128] . w + b); mode 0 = tanh, 1 = sigmoid
__global__ __launch_bounds__(256)
void k_head_gemv(const float* __restrict__ hin, const float* __restrict__ w,
                 const float* __restrict__ b, float* __restrict__ out, int mode) {
  int lane = threadIdx.x & 31, wv = threadIdx.x >> 5;
  int row = blockIdx.x * 8 + wv;
  const float* p = hin + (size_t)row * 128;
  float s = 0.f;
  #pragma unroll
  for (int q = 0; q < 4; ++q) s += p[q * 32 + lane] * w[q * 32 + lane];
  #pragma unroll
  for (int o = 16; o > 0; o >>= 1) s += __shfl_down(s, o, 32);
  if (lane == 0) {
    s += b[0];
    out[row] = (mode == 0) ? tanhf(s) : 1.f / (1.f + expf(-s));
  }
}

// ---------------------------------------------------------------------------
// Launch
// ---------------------------------------------------------------------------
extern "C" void kernel_launch(void* const* d_in, const int* in_sizes, int n_in,
                              void* d_out, int out_size, void* d_ws, size_t ws_size,
                              hipStream_t stream) {
  (void)in_sizes; (void)n_in; (void)out_size; (void)ws_size;

  const int*   nt    = (const int*)  d_in[IN_NODE_TYPES];
  const float* xrest = (const float*)d_in[IN_X_REST];
  const int*   eidx  = (const int*)  d_in[IN_EDGE_INDEX];
  const int*   src   = eidx;
  const int*   dst   = eidx + EE;
  const int*   eattr = (const int*)  d_in[IN_EDGE_ATTR];
  const float* gf    = (const float*)d_in[IN_GLOBAL_FEATS];
  const float* embN  = (const float*)d_in[IN_EMB_NODE];
  const float* embEd = (const float*)d_in[IN_EMB_EDGE];

  float* out = (float*)d_out;
  float* value_out  = out;
  float* policy_out = out + NG;
  float* tact_out   = out + NG + (size_t)NG * POLICY;
  float* strat_out  = tact_out + NG;

  // Workspace carving (float units, 16B-aligned regions)
  float* ws = (float*)d_ws;
  size_t o = 0;
  auto alloc = [&](size_t nf) {
    o = (o + 3) & ~(size_t)3;
    float* p = ws + o; o += nf; return p;
  };
  auto allocH = [&](size_t nh) { return (_Float16*)alloc((nh + 1) / 2); };

  float*    B0    = alloc((size_t)NN * HC);        // xl1 -> out2cat
  float*    B1    = alloc((size_t)NN * HC);        // xr1 -> xl2
  float*    B2    = alloc((size_t)NN * HC);        // out1cat -> xr2
  float*    S     = alloc((size_t)EE * HH);        // scores -> alpha (in place)
  _Float16* BXh   = allocH((size_t)NN * GAT_INP);  // padded node features, f16
  _Float16* X1h   = allocH((size_t)NN * HC);       // layer-1 output, f16
  _Float16* W1LT  = allocH((size_t)HC * GAT_INP);  // transposed f16 weights
  _Float16* W1RT  = allocH((size_t)HC * GAT_INP);
  _Float16* W2LT  = allocH((size_t)HC * HC);
  _Float16* W2RT  = allocH((size_t)HC * HC);
  _Float16* WV1T  = allocH((size_t)128 * MLP_IN);
  _Float16* WP1T  = allocH((size_t)256 * MLP_IN);
  _Float16* WT1T  = allocH((size_t)128 * MLP_IN);
  _Float16* WS1T  = allocH((size_t)128 * MLP_IN);
  _Float16* WP2T  = allocH((size_t)POLICY * 256);
  float*    EET1  = alloc(3 * HC);
  float*    EET2  = alloc(3 * HC);
  float*    GV    = alloc((size_t)NG * GEMB);
  _Float16* COMBh = allocH((size_t)NG * MLP_IN);
  float*    HV    = alloc((size_t)NG * 128);
  _Float16* HPh   = allocH((size_t)NG * 256);
  float*    HT    = alloc((size_t)NG * 128);
  float*    HS    = alloc((size_t)NG * 128);
  float*    X2    = alloc((size_t)NN * CC);

  auto gemm = [&](const _Float16* A, int lda, const _Float16* WT, int ldwt,
                  const float* bias, void* Out, int ldo,
                  int M, int Nn, int K, int act, int outHalf) {
    dim3 grid(Nn / 64, M / 128);
    k_wmma_gemm<<<grid, 256, 0, stream>>>(A, lda, WT, ldwt, bias, Out, ldo,
                                          M, Nn, K, act, outHalf);
  };
  auto prep = [&](int pidx, int K, int Kpad, int Nn, _Float16* WT) {
    k_prep_wt<<<((size_t)Nn * Kpad + 255) / 256, 256, 0, stream>>>(
        (const float*)d_in[pidx], K, Kpad, Nn, WT);
  };

  // ---- Preprocessing ----
  k_build_xh<<<(NN * GAT_INP + 255) / 256, 256, 0, stream>>>(nt, xrest, embN, BXh);
  prep(IN_G1WL, GAT_IN, GAT_INP, HC, W1LT);
  prep(IN_G1WR, GAT_IN, GAT_INP, HC, W1RT);
  prep(IN_G2WL, HC, HC, HC, W2LT);
  prep(IN_G2WR, HC, HC, HC, W2RT);
  prep(IN_WV1, MLP_IN, MLP_IN, 128, WV1T);
  prep(IN_WP1, MLP_IN, MLP_IN, 256, WP1T);
  prep(IN_WT1, MLP_IN, MLP_IN, 128, WT1T);
  prep(IN_WS1, MLP_IN, MLP_IN, 128, WS1T);
  prep(IN_WP2, 256, 256, POLICY, WP2T);
  k_edge_tab<<<3, HC, 0, stream>>>(embEd, (const float*)d_in[IN_G1WE], EET1);
  k_edge_tab<<<3, HC, 0, stream>>>(embEd, (const float*)d_in[IN_G2WE], EET2);
  k_global_mlp<<<NG, GEMB, 0, stream>>>(gf, (const float*)d_in[IN_WG],
                                        (const float*)d_in[IN_BG], GV);

  // ---- GAT layer 1 ----
  gemm(BXh, GAT_INP, W1LT, GAT_INP, (const float*)d_in[IN_G1BL], B0, HC, NN, HC, GAT_INP, 0, 0); // xl1
  gemm(BXh, GAT_INP, W1RT, GAT_INP, (const float*)d_in[IN_G1BR], B1, HC, NN, HC, GAT_INP, 0, 0); // xr1
  k_edge_scores<<<EE / 8, 256, 0, stream>>>(B0, B1, EET1, src, dst, eattr,
                                            (const float*)d_in[IN_G1ATT], S);
  k_softmax_graph<<<NG, EPG, 0, stream>>>(S, dst);
  k_aggregate<<<dim3(NG, HH), 256, 0, stream>>>(B0, S, src, dst, B2);   // out1cat
  k_ln_relu_512<<<NN, 256, 0, stream>>>(B2, (const float*)d_in[IN_G1BIAS],
                                        (const float*)d_in[IN_GAMMA1],
                                        (const float*)d_in[IN_BETA1], X1h);

  // ---- GAT layer 2 ----
  gemm(X1h, HC, W2LT, HC, (const float*)d_in[IN_G2BL], B1, HC, NN, HC, HC, 0, 0); // xl2
  gemm(X1h, HC, W2RT, HC, (const float*)d_in[IN_G2BR], B2, HC, NN, HC, HC, 0, 0); // xr2
  k_edge_scores<<<EE / 8, 256, 0, stream>>>(B1, B2, EET2, src, dst, eattr,
                                            (const float*)d_in[IN_G2ATT], S);
  k_softmax_graph<<<NG, EPG, 0, stream>>>(S, dst);
  k_aggregate<<<dim3(NG, HH), 256, 0, stream>>>(B1, S, src, dst, B0);   // out2cat
  k_ln2_relu<<<NN, CC, 0, stream>>>(B0, (const float*)d_in[IN_G2BIAS],
                                    (const float*)d_in[IN_GAMMA2],
                                    (const float*)d_in[IN_BETA2], X2);

  // ---- Pool + heads ----
  k_pool_concat<<<NG, CC, 0, stream>>>(X2, GV, COMBh);
  gemm(COMBh, MLP_IN, WV1T, MLP_IN, (const float*)d_in[IN_BV1], HV,  128, NG, 128, MLP_IN, 1, 0);
  gemm(COMBh, MLP_IN, WP1T, MLP_IN, (const float*)d_in[IN_BP1], HPh, 256, NG, 256, MLP_IN, 1, 1);
  gemm(COMBh, MLP_IN, WT1T, MLP_IN, (const float*)d_in[IN_BT1], HT,  128, NG, 128, MLP_IN, 1, 0);
  gemm(COMBh, MLP_IN, WS1T, MLP_IN, (const float*)d_in[IN_BS1], HS,  128, NG, 128, MLP_IN, 1, 0);
  gemm(HPh, 256, WP2T, 256, (const float*)d_in[IN_BP2], policy_out, POLICY,
       NG, POLICY, 256, 0, 0);
  k_head_gemv<<<NG / 8, 256, 0, stream>>>(HV, (const float*)d_in[IN_WV2],
                                          (const float*)d_in[IN_BV2], value_out, 0);
  k_head_gemv<<<NG / 8, 256, 0, stream>>>(HT, (const float*)d_in[IN_WT2],
                                          (const float*)d_in[IN_BT2], tact_out, 1);
  k_head_gemv<<<NG / 8, 256, 0, stream>>>(HS, (const float*)d_in[IN_WS2],
                                          (const float*)d_in[IN_BS2], strat_out, 1);
}
